// FlashLlamaAttention_40587440947292
// MI455X (gfx1250) — matrix operand order repaired and checked by reading, not compiled, not run on hardware
//
#include <hip/hip_runtime.h>

// ---------------------------------------------------------------------------
// Problem constants (match reference)
// ---------------------------------------------------------------------------
constexpr int HIDDEN   = 4096;
constexpr int NHEADS   = 32;
constexpr int HD       = 128;          // head dim
constexpr int BATCH    = 4;
constexpr int SEQ      = 1024;
constexpr int TOTAL    = BATCH * SEQ;  // 4096
constexpr float SOFTMAX_SCALE = 0.08838834764831845f; // 128^-0.5

typedef __attribute__((ext_vector_type(16))) __bf16 v16bf;
typedef __attribute__((ext_vector_type(8)))  __bf16 v8bf;
typedef __attribute__((ext_vector_type(4)))  __bf16 v4bf;
typedef __attribute__((ext_vector_type(8)))  float  v8f;
typedef __attribute__((ext_vector_type(4)))  float  v4f;

// ---------------------------------------------------------------------------
// Helpers
// ---------------------------------------------------------------------------
__device__ __forceinline__ __bf16 f2bf(float x) {
    union { float f; unsigned u; } v; v.f = x;
    unsigned r = v.u + 0x7FFFu + ((v.u >> 16) & 1u);   // round-to-nearest-even
    unsigned short h = (unsigned short)(r >> 16);
    return __builtin_bit_cast(__bf16, h);
}
__device__ __forceinline__ float bf2f(__bf16 x) {
    unsigned short h = __builtin_bit_cast(unsigned short, x);
    unsigned u = ((unsigned)h) << 16;
    return __builtin_bit_cast(float, u);
}

__device__ __forceinline__ v16bf cat8(v8bf lo, v8bf hi) {
    return __builtin_shufflevector(lo, hi, 0,1,2,3,4,5,6,7,8,9,10,11,12,13,14,15);
}

// Epilogue store: overload resolves the output type
__device__ __forceinline__ void store_out(float* C, size_t idx, float v)  { C[idx] = v; }
__device__ __forceinline__ void store_out(__bf16* C, size_t idx, float v) { C[idx] = f2bf(v); }

// D = A(16x32 bf16) x B(32x16 bf16) + C, fp32 accum
__device__ __forceinline__ v8f wmma_bf16(v16bf a, v16bf b, v8f c) {
    return __builtin_amdgcn_wmma_f32_16x16x32_bf16(
        /*neg_a=*/false, a, /*neg_b=*/false, b,
        /*c_mod=*/(short)0, c, /*reuse_a=*/false, /*reuse_b=*/false);
}

// A-fragment 16x32 (MxK). Lane l holds row (l&15).
// Lanes 0-15: K=0..7 and 16..23 ; lanes 16-31: K=8..15 and 24..31.
__device__ __forceinline__ v16bf load_afrag(const __bf16* p, int lda, int lane) {
    int r = lane & 15, half = lane >> 4;
    const __bf16* q = p + (size_t)r * lda + half * 8;
    return cat8(*(const v8bf*)q, *(const v8bf*)(q + 16));
}

// B-fragment 32x16 (KxN) for C[m][n]=sum_k A[m][k]*B'[n][k], B' row-major [N,K].
// Lane l holds column (l&15); lanes 0-15: K=0..7 and 16..23 ; lanes 16-31: K=8..15 and 24..31.
__device__ __forceinline__ v16bf load_bfrag(const __bf16* p, int ldb, int lane) {
    int c = lane & 15, half = lane >> 4;
    const __bf16* q = p + (size_t)c * ldb + half * 8;
    return cat8(*(const v8bf*)q, *(const v8bf*)(q + 16));
}

// Async DMA: 16B per lane, global -> LDS (tracked by ASYNCcnt).
// lds_off is the wave-relative LDS byte offset (low 32 bits of generic addr).
__device__ __forceinline__ void async_copy16(unsigned lds_off, const __bf16* g) {
    asm volatile("global_load_async_to_lds_b128 %0, %1, off"
                 :: "v"(lds_off), "v"(g) : "memory");
}
__device__ __forceinline__ void wait_async0() {
    asm volatile("s_wait_asynccnt 0" ::: "memory");
}

// ---------------------------------------------------------------------------
// fp32 -> bf16 cast (vectorized, bandwidth-bound pass)
// ---------------------------------------------------------------------------
__global__ __launch_bounds__(256) void cast4_kernel(const float* __restrict__ in,
                                                    __bf16* __restrict__ out,
                                                    size_t n4) {
    size_t i = (size_t)blockIdx.x * blockDim.x + threadIdx.x;
    if (i >= n4) return;
    v4f a = *(const v4f*)(in + 4 * i);
    v4bf r;
    r[0] = f2bf(a[0]); r[1] = f2bf(a[1]); r[2] = f2bf(a[2]); r[3] = f2bf(a[3]);
    *(v4bf*)(out + 4 * i) = r;
}

// ---------------------------------------------------------------------------
// bf16 WMMA GEMM with async-LDS double buffering:
//   C[M,N] = A[M,K] * B[N,K]^T
// Block = 128 threads = 4 waves (2x2), block tile 128x128, K-step 32.
// A/B tiles (128x32 bf16, 80B padded rows) staged via GLOBAL_LOAD_ASYNC_TO_LDS,
// next tile's DMA overlapped with current tile's 64 WMMAs.
// ---------------------------------------------------------------------------
template <typename OutT>
__global__ __launch_bounds__(128) void gemm_bt_kernel(const __bf16* __restrict__ A,
                                                      const __bf16* __restrict__ B,
                                                      OutT* __restrict__ C,
                                                      int M, int N, int K) {
    __shared__ __bf16 abuf[2][128][40];   // 80B row stride: conflict-free b128
    __shared__ __bf16 bbuf[2][128][40];

    const int tid  = threadIdx.x;
    const int lane = tid & 31;
    const int wave = tid >> 5;
    const int rowblk = blockIdx.x * 128;
    const int colblk = blockIdx.y * 128;
    const int wrow = (wave >> 1) * 64;
    const int wcol = (wave & 1) * 64;

    // Wave-relative LDS byte offsets of this thread's staging rows
    const unsigned a_lds0 = (unsigned)(unsigned long long)&abuf[0][tid][0];
    const unsigned b_lds0 = (unsigned)(unsigned long long)&bbuf[0][tid][0];
    const unsigned bufstride = 128u * 40u * (unsigned)sizeof(__bf16); // 10240 B

    // Each thread DMAs one A row and one B row of the block tile per K-step
    const __bf16* ag = A + (size_t)(rowblk + tid) * K;
    const __bf16* bg = B + (size_t)(colblk + tid) * K;

    auto issue_tile = [&](int buf, int k0) {
        unsigned al = a_lds0 + (unsigned)buf * bufstride;
        unsigned bl = b_lds0 + (unsigned)buf * bufstride;
        const __bf16* ap = ag + k0;
        const __bf16* bp = bg + k0;
#pragma unroll
        for (int c = 0; c < 4; ++c) {
            async_copy16(al + (unsigned)c * 16u, ap + c * 8);
            async_copy16(bl + (unsigned)c * 16u, bp + c * 8);
        }
    };

    v8f acc[4][4] = {};

    issue_tile(0, 0);

    for (int k0 = 0; k0 < K; k0 += 32) {
        const int buf = (k0 >> 5) & 1;
        wait_async0();        // this wave's DMA for `buf` complete
        __syncthreads();      // all waves' DMA visible; prior reads of buf^1 done
        if (k0 + 32 < K) issue_tile(buf ^ 1, k0 + 32);   // overlap next DMA

        v16bf af[4];
#pragma unroll
        for (int i = 0; i < 4; ++i)
            af[i] = load_afrag(&abuf[buf][wrow + 16 * i][0], 40, lane);
#pragma unroll
        for (int j = 0; j < 4; ++j) {
            v16bf bfr = load_bfrag(&bbuf[buf][wcol + 16 * j][0], 40, lane);
#pragma unroll
            for (int i = 0; i < 4; ++i)
                acc[i][j] = wmma_bf16(af[i], bfr, acc[i][j]);
        }
    }

    // C/D layout: VGPR j -> M = j + 8*(lane>=16), N = lane&15
    const int ncol  = lane & 15;
    const int mhalf = (lane >> 4) * 8;
#pragma unroll
    for (int i = 0; i < 4; ++i)
#pragma unroll
        for (int j = 0; j < 4; ++j)
#pragma unroll
            for (int e = 0; e < 8; ++e) {
                int m = rowblk + wrow + 16 * i + mhalf + e;
                int n = colblk + wcol + 16 * j + ncol;
                store_out(C, (size_t)m * N + n, acc[i][j][e]);
            }
}

// ---------------------------------------------------------------------------
// RoPE + layout shuffle.
// qkv: bf16 [T, 3*HIDDEN] ; outputs:
//   q,k : bf16 [B*H, S, D]   (row-major over d, WMMA A/B friendly)
//   vt  : bf16 [B*H, D, S]   (V transposed so PV B-operand is contiguous)
// ---------------------------------------------------------------------------
__global__ __launch_bounds__(256) void rope_kernel(const __bf16* __restrict__ qkv,
                                                   const float* __restrict__ cosT,
                                                   const float* __restrict__ sinT,
                                                   __bf16* __restrict__ qout,
                                                   __bf16* __restrict__ kout,
                                                   __bf16* __restrict__ vt) {
    int idx = blockIdx.x * 256 + threadIdx.x;
    if (idx >= TOTAL * NHEADS * (HD / 2)) return;
    int i = idx & 63;
    int h = (idx >> 6) & 31;
    int t = idx >> 11;
    int b = t >> 10, s = t & 1023;

    size_t base = (size_t)t * (3 * HIDDEN) + (size_t)h * HD;
    float c  = cosT[t * 64 + i];
    float sn = sinT[t * 64 + i];

    float q1 = bf2f(qkv[base + i]);
    float q2 = bf2f(qkv[base + 64 + i]);
    float k1 = bf2f(qkv[base + HIDDEN + i]);
    float k2 = bf2f(qkv[base + HIDDEN + 64 + i]);

    int bh = b * NHEADS + h;
    size_t qkbase = ((size_t)bh * SEQ + s) * HD;
    qout[qkbase + i]      = f2bf(q1 * c - q2 * sn);
    qout[qkbase + 64 + i] = f2bf(q2 * c + q1 * sn);
    kout[qkbase + i]      = f2bf(k1 * c - k2 * sn);
    kout[qkbase + 64 + i] = f2bf(k2 * c + k1 * sn);

    size_t vtb = (size_t)bh * HD * SEQ;
    vt[vtb + (size_t)i * SEQ + s]        = qkv[base + 2 * HIDDEN + i];
    vt[vtb + (size_t)(i + 64) * SEQ + s] = qkv[base + 2 * HIDDEN + 64 + i];
}

// ---------------------------------------------------------------------------
// Causal flash attention (online softmax), all matmuls via WMMA bf16.
// Grid: (SEQ/64, B*H). Block: 128 threads = 4 waves; each wave owns a 16-row
// Q tile. K-blocks of 32 keys: 8 QK WMMAs + 8 PV WMMAs per block.
// ---------------------------------------------------------------------------
__global__ __launch_bounds__(128) void attn_kernel(const __bf16* __restrict__ Q,
                                                   const __bf16* __restrict__ Kk,
                                                   const __bf16* __restrict__ Vt,
                                                   __bf16* __restrict__ O) {
    // Per-wave P staging tile, 80B row stride => conflict-free b128 reads.
    __shared__ __bf16 plds[4][16][40];

    const int lane = threadIdx.x & 31;
    const int wave = threadIdx.x >> 5;
    const int bh   = blockIdx.y;
    const int b    = bh >> 5;
    const int h    = bh & 31;
    const int q0   = blockIdx.x * 64 + wave * 16;   // first Q row of this wave

    const __bf16* Qb = Q  + ((size_t)bh * SEQ + q0) * HD;
    const __bf16* Kb = Kk + (size_t)bh * SEQ * HD;
    const __bf16* Vb = Vt + (size_t)bh * HD * SEQ;

    // Q fragments for the whole d=128, kept in registers (4 x 16x32)
    v16bf qf[4];
#pragma unroll
    for (int dc = 0; dc < 4; ++dc)
        qf[dc] = load_afrag(Qb + dc * 32, HD, lane);

    v8f acc[8] = {};            // 16x128 output accumulator (8 d-chunks)
    float mrow[8], lrow[8];
#pragma unroll
    for (int j = 0; j < 8; ++j) { mrow[j] = -1e30f; lrow[j] = 0.0f; }

    const int ncol  = lane & 15;
    const int rhalf = (lane >> 4) * 8;

    for (int kb = 0; kb <= q0 + 15; kb += 32) {
        // ---- scores: S = Q * K^T for 32 keys (two 16x16 tiles) ----
        v8f s0 = {}, s1 = {};
#pragma unroll
        for (int dc = 0; dc < 4; ++dc) {
            v16bf k0f = load_bfrag(Kb + (size_t)kb * HD + dc * 32, HD, lane);
            v16bf k1f = load_bfrag(Kb + (size_t)(kb + 16) * HD + dc * 32, HD, lane);
            s0 = wmma_bf16(qf[dc], k0f, s0);
            s1 = wmma_bf16(qf[dc], k1f, s1);
        }

        // ---- causal mask + online softmax update (rows live in 16-lane halves)
#pragma unroll
        for (int j = 0; j < 8; ++j) {
            int row = q0 + rhalf + j;
            float a0 = (kb + ncol      <= row) ? s0[j] * SOFTMAX_SCALE : -1e30f;
            float a1 = (kb + 16 + ncol <= row) ? s1[j] * SOFTMAX_SCALE : -1e30f;
            float mx = fmaxf(a0, a1);
#pragma unroll
            for (int msk = 1; msk < 16; msk <<= 1)
                mx = fmaxf(mx, __shfl_xor(mx, msk, 32));
            float mnew = fmaxf(mrow[j], mx);
            float p0 = __expf(a0 - mnew);
            float p1 = __expf(a1 - mnew);
            float ps = p0 + p1;
#pragma unroll
            for (int msk = 1; msk < 16; msk <<= 1)
                ps += __shfl_xor(ps, msk, 32);
            float corr = __expf(mrow[j] - mnew);
            lrow[j] = lrow[j] * corr + ps;
            mrow[j] = mnew;
#pragma unroll
            for (int n = 0; n < 8; ++n) acc[n][j] *= corr;
            plds[wave][rhalf + j][ncol]      = f2bf(p0);
            plds[wave][rhalf + j][16 + ncol] = f2bf(p1);
        }

        // ---- reload P in A-fragment layout from LDS (same wave, in-order DS)
        {
            int r = lane & 15, half = lane >> 4;
            const __bf16* p = &plds[wave][r][half * 8];
            v16bf pf = cat8(*(const v8bf*)p, *(const v8bf*)(p + 16));
            // ---- O += P * V  (V^T rows contiguous => B-fragment loads)
#pragma unroll
            for (int n = 0; n < 8; ++n) {
                v16bf vf = load_bfrag(Vb + (size_t)(n * 16) * SEQ + kb, SEQ, lane);
                acc[n] = wmma_bf16(pf, vf, acc[n]);
            }
        }
    }

    // ---- epilogue: normalize and scatter into attn[T, HIDDEN] (bf16)
#pragma unroll
    for (int j = 0; j < 8; ++j) {
        float inv = 1.0f / lrow[j];
        int t = b * SEQ + q0 + rhalf + j;
        size_t orow = (size_t)t * HIDDEN + (size_t)h * HD;
#pragma unroll
        for (int n = 0; n < 8; ++n)
            O[orow + n * 16 + ncol] = f2bf(acc[n][j] * inv);
    }
}

// ---------------------------------------------------------------------------
// Host orchestration
// ---------------------------------------------------------------------------
extern "C" void kernel_launch(void* const* d_in, const int* in_sizes, int n_in,
                              void* d_out, int out_size, void* d_ws, size_t ws_size,
                              hipStream_t stream) {
    const float* hidden = (const float*)d_in[0];
    const float* cosT   = (const float*)d_in[1];
    const float* sinT   = (const float*)d_in[2];
    const float* wqkv   = (const float*)d_in[3];
    const float* wo     = (const float*)d_in[4];
    float* out = (float*)d_out;

    char* ws = (char*)d_ws;
    size_t off = 0;
    auto alloc = [&](size_t bytes) -> __bf16* {
        char* p = ws + off;
        off += (bytes + 255) & ~(size_t)255;
        return (__bf16*)p;
    };
    __bf16* wqkv_bf = alloc((size_t)3 * HIDDEN * HIDDEN * 2);   // 100.7 MB
    __bf16* wo_bf   = alloc((size_t)HIDDEN * HIDDEN * 2);       //  33.6 MB
    __bf16* hid_bf  = alloc((size_t)TOTAL * HIDDEN * 2);        //  33.6 MB
    __bf16* q_bf    = alloc((size_t)TOTAL * HIDDEN * 2);
    __bf16* k_bf    = alloc((size_t)TOTAL * HIDDEN * 2);
    __bf16* vt_bf   = alloc((size_t)TOTAL * HIDDEN * 2);
    __bf16* qkv_bf  = alloc((size_t)TOTAL * 3 * HIDDEN * 2);    // 100.7 MB
    __bf16* attn_bf = hid_bf;  // hidden_bf dead after QKV GEMM -> reuse

    // 1) casts to bf16
    {
        size_t n4;
        n4 = (size_t)3 * HIDDEN * HIDDEN / 4;
        cast4_kernel<<<(int)((n4 + 255) / 256), 256, 0, stream>>>(wqkv, wqkv_bf, n4);
        n4 = (size_t)HIDDEN * HIDDEN / 4;
        cast4_kernel<<<(int)((n4 + 255) / 256), 256, 0, stream>>>(wo, wo_bf, n4);
        n4 = (size_t)TOTAL * HIDDEN / 4;
        cast4_kernel<<<(int)((n4 + 255) / 256), 256, 0, stream>>>(hidden, hid_bf, n4);
    }

    // 2) qkv = hidden @ wqkv^T   [4096 x 12288], bf16 out
    {
        dim3 grid(TOTAL / 128, (3 * HIDDEN) / 128);
        gemm_bt_kernel<__bf16><<<grid, 128, 0, stream>>>(
            hid_bf, wqkv_bf, qkv_bf, TOTAL, 3 * HIDDEN, HIDDEN);
    }

    // 3) RoPE + split/transpose into q,k [BH,S,D] and v^T [BH,D,S]
    {
        int n = TOTAL * NHEADS * (HD / 2);
        rope_kernel<<<(n + 255) / 256, 256, 0, stream>>>(
            qkv_bf, cosT, sinT, q_bf, k_bf, vt_bf);
    }

    // 4) causal flash attention -> attn_bf [T, HIDDEN]
    {
        dim3 grid(SEQ / 64, BATCH * NHEADS);
        attn_kernel<<<grid, 128, 0, stream>>>(q_bf, k_bf, vt_bf, attn_bf);
    }

    // 5) out = attn @ wo^T  (fp32 result)
    {
        dim3 grid(TOTAL / 128, HIDDEN / 128);
        gemm_bt_kernel<float><<<grid, 128, 0, stream>>>(
            attn_bf, wo_bf, out, TOTAL, HIDDEN, HIDDEN);
    }
}
